// PointNet2BackboneLight_20220706029883
// MI455X (gfx1250) — matrix operand order repaired and checked by reading, not compiled, not run on hardware
//
#include <hip/hip_runtime.h>
#include <hip/hip_bf16.h>

#define BB   4
#define NPER 20000
#define NS   16
#define CDIV(a,b) (((a)+(b)-1)/(b))

typedef __attribute__((ext_vector_type(16))) _Float16 v16h;
typedef __attribute__((ext_vector_type(8)))  _Float16 h8;
typedef __attribute__((ext_vector_type(8)))  float    v8f;

// ---------------------------------------------------------------------------
// coords (int4 per point) -> xyz f32
__global__ void k_xyz(const int* __restrict__ coords, float* __restrict__ xyz, int total)
{
  int i = blockIdx.x * blockDim.x + threadIdx.x;
  if (i >= total) return;
  xyz[i*3+0] = (float)coords[i*4+1] * 0.005f;
  xyz[i*3+1] = (float)coords[i*4+2] * 0.005f;
  xyz[i*3+2] = (float)coords[i*4+3] * 0.005f;
}

// ---------------------------------------------------------------------------
// fold BN into weights: Wh[n][k] f16 (zero padded to kpad), scale/shift f32
__global__ void k_prep(const float* __restrict__ W, const float* __restrict__ bias,
                       const float* __restrict__ gamma, const float* __restrict__ beta,
                       _Float16* __restrict__ Wh, float* __restrict__ scale,
                       float* __restrict__ shift, int cout, int cin, int kpad)
{
  int id = blockIdx.x * blockDim.x + threadIdx.x;
  if (id < cout * kpad) {
    int n = id / kpad, k = id - n * kpad;
    Wh[id] = (k < cin) ? (_Float16)W[n*cin + k] : (_Float16)0.0f;
  }
  if (id < cout) {
    float s = gamma[id] * 0.9999950000374997f; // 1/sqrt(1+1e-5)
    scale[id] = s;
    shift[id] = bias[id] * s + beta[id];
  }
}

// ---------------------------------------------------------------------------
// farthest point sampling: 1 block per batch, sequential npoint steps,
// LDS argmax reduction, persistent dist in global scratch
__global__ void k_fps(const float* __restrict__ xyz, int* __restrict__ idx,
                      float* __restrict__ dist, int N, int npoint)
{
  int b = blockIdx.x;
  const float* p = xyz + (size_t)b * N * 3;
  float* d = dist + (size_t)b * N;
  __shared__ float sval[256];
  __shared__ int   sidx[256];
  int t = threadIdx.x;
  for (int i = t; i < N; i += 256) d[i] = 1e10f;
  __syncthreads();
  int last = 0;
  for (int s = 0; s < npoint; ++s) {
    if (t == 0) idx[(size_t)b * npoint + s] = last;
    float cx = p[last*3+0], cy = p[last*3+1], cz = p[last*3+2];
    float best = -1.0f; int bi = 0;
    for (int i = t; i < N; i += 256) {
      float dx = p[i*3+0]-cx, dy = p[i*3+1]-cy, dz = p[i*3+2]-cz;
      float dd = dx*dx + dy*dy + dz*dz;
      float old = d[i];
      float nv  = old < dd ? old : dd;
      d[i] = nv;
      if (nv > best) { best = nv; bi = i; }
    }
    sval[t] = best; sidx[t] = bi;
    __syncthreads();
    for (int off = 128; off > 0; off >>= 1) {
      if (t < off && sval[t+off] > sval[t]) { sval[t] = sval[t+off]; sidx[t] = sidx[t+off]; }
      __syncthreads();
    }
    last = sidx[0];
    __syncthreads();
  }
}

// ---------------------------------------------------------------------------
// dst[b,i,c] = src[b, idx[b,i], c]
__global__ void k_gather(const float* __restrict__ src, const int* __restrict__ idx,
                         float* __restrict__ dst, int Bn, int N, int S, int C)
{
  int id = blockIdx.x * blockDim.x + threadIdx.x;
  if (id >= Bn * S * C) return;
  int c = id % C; int r = id / C; int b = r / S;
  int j = idx[r];
  dst[id] = src[((size_t)b * N + j) * C + c];
}

// ---------------------------------------------------------------------------
// ball query: first ns indices with d2 < r2, pad with first (or 0 if none)
__global__ void k_ball(const float* __restrict__ xyz, const float* __restrict__ nx,
                       int* __restrict__ gidx, int Bn, int N, int S, int ns, float r2)
{
  int id = blockIdx.x * blockDim.x + threadIdx.x;
  if (id >= Bn * S) return;
  int b = id / S;
  const float* p = xyz + (size_t)b * N * 3;
  float cx = nx[(size_t)id*3+0], cy = nx[(size_t)id*3+1], cz = nx[(size_t)id*3+2];
  int* out = gidx + (size_t)id * ns;
  int cnt = 0, first = -1;
  for (int j = 0; j < N; ++j) {
    float dx = p[j*3+0]-cx, dy = p[j*3+1]-cy, dz = p[j*3+2]-cz;
    float dd = dx*dx + dy*dy + dz*dz;
    if (dd < r2) {
      if (first < 0) first = j;
      out[cnt++] = j;
      if (cnt == ns) break;
    }
  }
  if (first < 0) { for (int k = 0;  k < ns; ++k) out[k] = 0; }
  else           { for (int k = cnt; k < ns; ++k) out[k] = first; }
}

// ---------------------------------------------------------------------------
// build grouped f16 GEMM input: [rel_xyz | feats | 0-pad] per (b,i,s)
__global__ void k_group(const float* __restrict__ xyz, const float* __restrict__ nx,
                        const float* __restrict__ feats, const int* __restrict__ gidx,
                        _Float16* __restrict__ Ah, int Bn, int N, int S, int ns,
                        int C, int Kpad)
{
  int id = blockIdx.x * blockDim.x + threadIdx.x;
  if (id >= Bn * S * ns) return;
  int bi = id / ns;
  int b  = bi / S;
  int j  = gidx[id];
  const float* p = xyz + ((size_t)b * N + j) * 3;
  const float* c = nx + (size_t)bi * 3;
  _Float16* o = Ah + (size_t)id * Kpad;
  o[0] = (_Float16)(p[0] - c[0]);
  o[1] = (_Float16)(p[1] - c[1]);
  o[2] = (_Float16)(p[2] - c[2]);
  const float* f = feats + ((size_t)b * N + j) * C;
  for (int k = 0; k < C; ++k)        o[3+k] = (_Float16)f[k];
  for (int k = 3 + C; k < Kpad; ++k) o[k]   = (_Float16)0.0f;
}

// ---------------------------------------------------------------------------
// WMMA GEMM + folded-BN epilogue.  out = relu?(A[M,K] x Wt[N,K]^T * s + c)
// One wave computes a 16x64 block: A fragment reused across 4 WMMA tiles.
// K % 32 == 0, M % 16 == 0, N % 64 == 0.
__global__ __launch_bounds__(32)
void k_gemm_bn(const _Float16* __restrict__ A, const _Float16* __restrict__ Wt,
               const float* __restrict__ scale, const float* __restrict__ shift,
               float* __restrict__ outF, _Float16* __restrict__ outH,
               int M, int N, int K, int relu)
{
  int lane   = threadIdx.x;
  int m0     = blockIdx.x * 16;
  int n0     = blockIdx.y * 64;
  int lanelo = lane & 15, lanehi = lane >> 4;

  v8f acc[4] = { v8f{}, v8f{}, v8f{}, v8f{} };
  const _Float16* arow = A  + (size_t)(m0 + lanelo) * K;
  const _Float16* brow = Wt + (size_t)(n0 + lanelo) * K;

  for (int k0 = 0; k0 < K; k0 += 32) {
    // A 16x32 f16 fragment: lane<16 -> K = e + (e>=8?8:0); lane>=16 -> +8
    const _Float16* ap = arow + k0 + 8 * lanehi;
    v16h a = __builtin_shufflevector(*(const h8*)ap, *(const h8*)(ap + 16),
                                     0,1,2,3,4,5,6,7,8,9,10,11,12,13,14,15);
    // B 32x16 f16 fragment: same K order as A (e + (e>=8?8:0), +8 for lane>=16); N = lane&15,
    // and B[:,n] == row n of Wt -> contiguous loads
    int bo = k0 + 8 * lanehi;
#pragma unroll
    for (int t = 0; t < 4; ++t) {
      const _Float16* bp = brow + (size_t)(16 * t) * K + bo;
      v16h bfrag = __builtin_shufflevector(*(const h8*)bp, *(const h8*)(bp + 16),
                                           0,1,2,3,4,5,6,7,8,9,10,11,12,13,14,15);
      acc[t] = __builtin_amdgcn_wmma_f32_16x16x32_f16(
                 false, a, false, bfrag, (short)0, acc[t], false, false);
    }
  }

#pragma unroll
  for (int t = 0; t < 4; ++t) {
    int n = n0 + 16 * t + lanelo;
    float sc = scale[n], sh = shift[n];
#pragma unroll
    for (int r = 0; r < 8; ++r) {
      int m = m0 + r + 8 * lanehi;
      float v = acc[t][r] * sc + sh;
      if (relu) v = fmaxf(v, 0.0f);
      size_t o = (size_t)m * N + n;
      if (outF) outF[o] = v;
      if (outH) outH[o] = (_Float16)v;
    }
  }
}

// ---------------------------------------------------------------------------
// max over the nsample axis: in [rows*ns, C] -> out [rows, C]
__global__ void k_maxpool(const float* __restrict__ in, float* __restrict__ out,
                          int rows, int ns, int C)
{
  int id = blockIdx.x * blockDim.x + threadIdx.x;
  if (id >= rows * C) return;
  int c = id % C; int r = id / C;
  const float* p = in + ((size_t)r * ns) * C + c;
  float m = p[0];
  for (int s = 1; s < ns; ++s) m = fmaxf(m, p[(size_t)s * C]);
  out[(size_t)r * C + c] = m;
}

// ---------------------------------------------------------------------------
// three nearest neighbors + inverse-distance weights
__global__ void k_three_nn(const float* __restrict__ ux, const float* __restrict__ kx,
                           int* __restrict__ idx3, float* __restrict__ w3,
                           int Bn, int Nu, int Nk)
{
  int id = blockIdx.x * blockDim.x + threadIdx.x;
  if (id >= Bn * Nu) return;
  int b = id / Nu;
  float px = ux[(size_t)id*3+0], py = ux[(size_t)id*3+1], pz = ux[(size_t)id*3+2];
  const float* kp = kx + (size_t)b * Nk * 3;
  float d0 = 1e30f, d1 = 1e30f, d2v = 1e30f;
  int   i0 = 0,     i1 = 0,     i2 = 0;
  for (int j = 0; j < Nk; ++j) {
    float dx = kp[j*3+0]-px, dy = kp[j*3+1]-py, dz = kp[j*3+2]-pz;
    float dd = dx*dx + dy*dy + dz*dz;
    if      (dd < d0)  { d2v=d1; i2=i1; d1=d0; i1=i0; d0=dd; i0=j; }
    else if (dd < d1)  { d2v=d1; i2=i1; d1=dd; i1=j; }
    else if (dd < d2v) { d2v=dd; i2=j; }
  }
  float r0 = 1.0f / (sqrtf(fmaxf(d0, 0.0f)) + 1e-8f);
  float r1 = 1.0f / (sqrtf(fmaxf(d1, 0.0f)) + 1e-8f);
  float r2 = 1.0f / (sqrtf(fmaxf(d2v,0.0f)) + 1e-8f);
  float s  = r0 + r1 + r2;
  idx3[(size_t)id*3+0] = i0; idx3[(size_t)id*3+1] = i1; idx3[(size_t)id*3+2] = i2;
  w3[(size_t)id*3+0] = r0/s; w3[(size_t)id*3+1] = r1/s; w3[(size_t)id*3+2] = r2/s;
}

// ---------------------------------------------------------------------------
// FP input builder: Ah[pt, c] = [interp(kf) | uf | 0-pad] in f16
__global__ void k_interp(const float* __restrict__ kf, const int* __restrict__ idx3,
                         const float* __restrict__ w3, const float* __restrict__ uf,
                         _Float16* __restrict__ Ah, int Bn, int Nu, int Nk,
                         int Ck, int Cu, int Kpad)
{
  long long id = (long long)blockIdx.x * blockDim.x + threadIdx.x;
  long long total = (long long)Bn * Nu * Kpad;
  if (id >= total) return;
  int c = (int)(id % Kpad);
  long long pt = id / Kpad;
  int b = (int)(pt / Nu);
  float v = 0.0f;
  if (c < Ck) {
    const int*   ii = idx3 + pt * 3;
    const float* ww = w3   + pt * 3;
    const float* base = kf + (size_t)b * Nk * Ck;
    v = ww[0] * base[(size_t)ii[0] * Ck + c]
      + ww[1] * base[(size_t)ii[1] * Ck + c]
      + ww[2] * base[(size_t)ii[2] * Ck + c];
  } else if (c < Ck + Cu) {
    v = uf[pt * Cu + (c - Ck)];
  }
  Ah[id] = (_Float16)v;
}

// ===========================================================================
extern "C" void kernel_launch(void* const* d_in, const int* in_sizes, int n_in,
                              void* d_out, int out_size, void* d_ws, size_t ws_size,
                              hipStream_t stream)
{
  (void)in_sizes; (void)n_in; (void)out_size; (void)ws_size;
  const int*   coords = (const int*)d_in[0];
  const float* feats  = (const float*)d_in[1];

  struct LayerDim { int cin, cout, kpad; };
  static const LayerDim LD[13] = {
    {6,64,32},{64,128,64},          // sa1
    {131,128,160},{128,256,128},    // sa2
    {259,256,288},{256,512,256},    // sa3
    {768,256,768},{256,256,256},    // fp3
    {384,128,384},{128,128,128},    // fp2
    {131,128,160},{128,128,128},    // fp1
    {128,512,128}                   // final
  };
  const float *Wp[13], *bp[13], *gp[13], *betap[13];
  for (int l = 0; l < 13; ++l) {
    Wp[l]    = (const float*)d_in[2 + 4*l];
    bp[l]    = (const float*)d_in[3 + 4*l];
    gp[l]    = (const float*)d_in[4 + 4*l];
    betap[l] = (const float*)d_in[5 + 4*l];
  }

  size_t off = 0;
  auto alloc = [&](size_t bytes) -> void* {
    void* p = (char*)d_ws + off;
    off += (bytes + 255) & ~(size_t)255;
    return p;
  };

  // weights
  _Float16* Wh[13]; float* scl[13]; float* shf[13];
  for (int l = 0; l < 13; ++l) {
    Wh[l]  = (_Float16*)alloc((size_t)LD[l].cout * LD[l].kpad * 2);
    scl[l] = (float*)alloc((size_t)LD[l].cout * 4);
    shf[l] = (float*)alloc((size_t)LD[l].cout * 4);
  }
  // activations / scratch
  float* xyz   = (float*)alloc((size_t)BB*NPER*3*4);
  float* dist  = (float*)alloc((size_t)BB*NPER*4);
  int*   fidx0 = (int*)  alloc((size_t)BB*1024*4);
  float* xyzs  = (float*)alloc((size_t)BB*1024*3*4);
  float* l0f   = (float*)alloc((size_t)BB*1024*3*4);

  int*   fidx1 = (int*)  alloc((size_t)BB*256*4);
  float* l1x   = (float*)alloc((size_t)BB*256*3*4);
  int*   gidx1 = (int*)  alloc((size_t)BB*256*NS*4);
  _Float16* Asa1 = (_Float16*)alloc((size_t)BB*256*NS*32*2);
  _Float16* h1   = (_Float16*)alloc((size_t)BB*256*NS*64*2);
  float* m1f = (float*)alloc((size_t)BB*256*NS*128*4);
  float* l1  = (float*)alloc((size_t)BB*256*128*4);

  int*   fidx2 = (int*)  alloc((size_t)BB*64*4);
  float* l2x   = (float*)alloc((size_t)BB*64*3*4);
  int*   gidx2 = (int*)  alloc((size_t)BB*64*NS*4);
  _Float16* Asa2 = (_Float16*)alloc((size_t)BB*64*NS*160*2);
  _Float16* h2   = (_Float16*)alloc((size_t)BB*64*NS*128*2);
  float* m2f = (float*)alloc((size_t)BB*64*NS*256*4);
  float* l2  = (float*)alloc((size_t)BB*64*256*4);

  int*   fidx3 = (int*)  alloc((size_t)BB*16*4);
  float* l3x   = (float*)alloc((size_t)BB*16*3*4);
  int*   gidx3 = (int*)  alloc((size_t)BB*16*NS*4);
  _Float16* Asa3 = (_Float16*)alloc((size_t)BB*16*NS*288*2);
  _Float16* h3   = (_Float16*)alloc((size_t)BB*16*NS*256*2);
  float* m3f = (float*)alloc((size_t)BB*16*NS*512*4);
  float* l3  = (float*)alloc((size_t)BB*16*512*4);

  int* nn3i = (int*)alloc((size_t)BB*64*3*4);   float* nn3w = (float*)alloc((size_t)BB*64*3*4);
  _Float16* Af3 = (_Float16*)alloc((size_t)BB*64*768*2);
  _Float16* hf3 = (_Float16*)alloc((size_t)BB*64*256*2);
  float* l2n = (float*)alloc((size_t)BB*64*256*4);

  int* nn2i = (int*)alloc((size_t)BB*256*3*4);  float* nn2w = (float*)alloc((size_t)BB*256*3*4);
  _Float16* Af2 = (_Float16*)alloc((size_t)BB*256*384*2);
  _Float16* hf2 = (_Float16*)alloc((size_t)BB*256*128*2);
  float* l1n = (float*)alloc((size_t)BB*256*128*4);

  int* nn1i = (int*)alloc((size_t)BB*1024*3*4); float* nn1w = (float*)alloc((size_t)BB*1024*3*4);
  _Float16* Af1 = (_Float16*)alloc((size_t)BB*1024*160*2);
  _Float16* hf1 = (_Float16*)alloc((size_t)BB*1024*128*2);
  float* l0n = (float*)alloc((size_t)BB*1024*128*4);

  int* nnfi = (int*)alloc((size_t)BB*NPER*3*4); float* nnfw = (float*)alloc((size_t)BB*NPER*3*4);
  _Float16* uph = (_Float16*)alloc((size_t)BB*NPER*128*2);

  // ---- weight prep
  for (int l = 0; l < 13; ++l) {
    int tot = LD[l].cout * LD[l].kpad;
    k_prep<<<CDIV(tot,256),256,0,stream>>>(Wp[l],bp[l],gp[l],betap[l],
                                           Wh[l],scl[l],shf[l],
                                           LD[l].cout,LD[l].cin,LD[l].kpad);
  }

  // ---- points + initial FPS subsample 20000 -> 1024
  k_xyz<<<CDIV(BB*NPER,256),256,0,stream>>>(coords, xyz, BB*NPER);
  k_fps<<<BB,256,0,stream>>>(xyz, fidx0, dist, NPER, 1024);
  k_gather<<<CDIV(BB*1024*3,256),256,0,stream>>>(xyz,   fidx0, xyzs, BB, NPER, 1024, 3);
  k_gather<<<CDIV(BB*1024*3,256),256,0,stream>>>(feats, fidx0, l0f,  BB, NPER, 1024, 3);

  // ---- SA1: 1024 -> 256, r=0.04, cin 3+3
  k_fps<<<BB,256,0,stream>>>(xyzs, fidx1, dist, 1024, 256);
  k_gather<<<CDIV(BB*256*3,256),256,0,stream>>>(xyzs, fidx1, l1x, BB, 1024, 256, 3);
  k_ball<<<CDIV(BB*256,128),128,0,stream>>>(xyzs, l1x, gidx1, BB, 1024, 256, NS, 0.0016f);
  k_group<<<CDIV(BB*256*NS,256),256,0,stream>>>(xyzs, l1x, l0f, gidx1, Asa1, BB, 1024, 256, NS, 3, 32);
  k_gemm_bn<<<dim3(16384/16, 64/64), 32,0,stream>>>(Asa1, Wh[0], scl[0], shf[0], nullptr, h1, 16384, 64, 32, 1);
  k_gemm_bn<<<dim3(16384/16,128/64), 32,0,stream>>>(h1,   Wh[1], scl[1], shf[1], m1f, nullptr, 16384, 128, 64, 1);
  k_maxpool<<<CDIV(BB*256*128,256),256,0,stream>>>(m1f, l1, BB*256, NS, 128);

  // ---- SA2: 256 -> 64, r=0.08, cin 128+3
  k_fps<<<BB,256,0,stream>>>(l1x, fidx2, dist, 256, 64);
  k_gather<<<CDIV(BB*64*3,256),256,0,stream>>>(l1x, fidx2, l2x, BB, 256, 64, 3);
  k_ball<<<CDIV(BB*64,128),128,0,stream>>>(l1x, l2x, gidx2, BB, 256, 64, NS, 0.0064f);
  k_group<<<CDIV(BB*64*NS,256),256,0,stream>>>(l1x, l2x, l1, gidx2, Asa2, BB, 256, 64, NS, 128, 160);
  k_gemm_bn<<<dim3(4096/16,128/64), 32,0,stream>>>(Asa2, Wh[2], scl[2], shf[2], nullptr, h2, 4096, 128, 160, 1);
  k_gemm_bn<<<dim3(4096/16,256/64), 32,0,stream>>>(h2,   Wh[3], scl[3], shf[3], m2f, nullptr, 4096, 256, 128, 1);
  k_maxpool<<<CDIV(BB*64*256,256),256,0,stream>>>(m2f, l2, BB*64, NS, 256);

  // ---- SA3: 64 -> 16, r=0.16, cin 256+3
  k_fps<<<BB,256,0,stream>>>(l2x, fidx3, dist, 64, 16);
  k_gather<<<CDIV(BB*16*3,256),256,0,stream>>>(l2x, fidx3, l3x, BB, 64, 16, 3);
  k_ball<<<CDIV(BB*16,128),128,0,stream>>>(l2x, l3x, gidx3, BB, 64, 16, NS, 0.0256f);
  k_group<<<CDIV(BB*16*NS,256),256,0,stream>>>(l2x, l3x, l2, gidx3, Asa3, BB, 64, 16, NS, 256, 288);
  k_gemm_bn<<<dim3(1024/16,256/64), 32,0,stream>>>(Asa3, Wh[4], scl[4], shf[4], nullptr, h3, 1024, 256, 288, 1);
  k_gemm_bn<<<dim3(1024/16,512/64), 32,0,stream>>>(h3,   Wh[5], scl[5], shf[5], m3f, nullptr, 1024, 512, 256, 1);
  k_maxpool<<<CDIV(BB*16*512,256),256,0,stream>>>(m3f, l3, BB*16, NS, 512);

  // ---- FP3: interp l3 (512) onto l2_xyz, concat l2 (256) -> 768
  k_three_nn<<<CDIV(BB*64,128),128,0,stream>>>(l2x, l3x, nn3i, nn3w, BB, 64, 16);
  k_interp<<<CDIV(BB*64*768,256),256,0,stream>>>(l3, nn3i, nn3w, l2, Af3, BB, 64, 16, 512, 256, 768);
  k_gemm_bn<<<dim3(256/16,256/64), 32,0,stream>>>(Af3, Wh[6], scl[6], shf[6], nullptr, hf3, 256, 256, 768, 1);
  k_gemm_bn<<<dim3(256/16,256/64), 32,0,stream>>>(hf3, Wh[7], scl[7], shf[7], l2n, nullptr, 256, 256, 256, 1);

  // ---- FP2: interp l2n (256) onto l1_xyz, concat l1 (128) -> 384
  k_three_nn<<<CDIV(BB*256,128),128,0,stream>>>(l1x, l2x, nn2i, nn2w, BB, 256, 64);
  k_interp<<<CDIV(BB*256*384,256),256,0,stream>>>(l2n, nn2i, nn2w, l1, Af2, BB, 256, 64, 256, 128, 384);
  k_gemm_bn<<<dim3(1024/16,128/64), 32,0,stream>>>(Af2, Wh[8], scl[8], shf[8], nullptr, hf2, 1024, 128, 384, 1);
  k_gemm_bn<<<dim3(1024/16,128/64), 32,0,stream>>>(hf2, Wh[9], scl[9], shf[9], l1n, nullptr, 1024, 128, 128, 1);

  // ---- FP1: interp l1n (128) onto xyz_sub, concat l0_feats (3) -> 131 (pad 160)
  k_three_nn<<<CDIV(BB*1024,128),128,0,stream>>>(xyzs, l1x, nn1i, nn1w, BB, 1024, 256);
  k_interp<<<CDIV(BB*1024*160,256),256,0,stream>>>(l1n, nn1i, nn1w, l0f, Af1, BB, 1024, 256, 128, 3, 160);
  k_gemm_bn<<<dim3(4096/16,128/64), 32,0,stream>>>(Af1, Wh[10], scl[10], shf[10], nullptr, hf1, 4096, 128, 160, 1);
  k_gemm_bn<<<dim3(4096/16,128/64), 32,0,stream>>>(hf1, Wh[11], scl[11], shf[11], l0n, nullptr, 4096, 128, 128, 1);

  // ---- final: interp l0n (128) onto full 20000 pts, 128 -> 512 (no relu)
  k_three_nn<<<CDIV(BB*NPER,128),128,0,stream>>>(xyz, xyzs, nnfi, nnfw, BB, NPER, 1024);
  {
    long long tot = (long long)BB * NPER * 128;
    k_interp<<<(unsigned)CDIV(tot,256),256,0,stream>>>(l0n, nnfi, nnfw, nullptr, uph,
                                                       BB, NPER, 1024, 128, 0, 128);
  }
  k_gemm_bn<<<dim3(BB*NPER/16, 512/64), 32,0,stream>>>(uph, Wh[12], scl[12], shf[12],
                                                       (float*)d_out, nullptr,
                                                       BB*NPER, 512, 128, 0);
}
